// VQAE_65171833749600
// MI455X (gfx1250) — matrix operand order repaired and checked by reading, not compiled, not run on hardware
//
#include <hip/hip_runtime.h>
#include <hip/hip_bf16.h>
#include <math.h>

typedef __bf16 bf16;
typedef __attribute__((ext_vector_type(16))) __bf16 v16bf;
typedef __attribute__((ext_vector_type(8)))  __bf16 v8bf;
typedef __attribute__((ext_vector_type(8)))  float  v8f;
typedef __attribute__((ext_vector_type(4)))  unsigned int v4u;

#define EPS_BN 1e-5f
#define EPS_LN 1e-5f

__device__ __forceinline__ float silu_f(float v) { return v / (1.0f + expf(-v)); }
__device__ __forceinline__ int   iclampi(int v, int lo, int hi) { return v < lo ? lo : (v > hi ? hi : v); }

// Build a 16x32 bf16 A/B fragment half from two contiguous 8-element (16B) chunks.
// CDNA5 16-bit fragment layouts (ISA 7.12.2):
//   A: element e needs K = e + 8*half (e<8) | e + 8*half + 8 (e>=8)
//      -> chunks at kb + 8*half and kb + 16 + 8*half
//   B: same K order as A -> chunks at kb + 8*half and kb + 16 + 8*half
__device__ __forceinline__ v16bf frag16(const bf16* __restrict__ p0, const bf16* __restrict__ p1) {
  v8bf lo = *(const v8bf*)p0;
  v8bf hi = *(const v8bf*)p1;
  v16bf r;
  #pragma unroll
  for (int e = 0; e < 8; ++e) { r[e] = lo[e]; r[e + 8] = hi[e]; }
  return r;
}

// ---------------------------------------------------------------- utilities
__global__ void k_zero(float* __restrict__ p, int n) {
  int i = blockIdx.x * blockDim.x + threadIdx.x;
  if (i < n) p[i] = 0.0f;
}

// 16-byte-granular zero fill (n16 = number of 16B chunks)
__global__ void k_memset16(v4u* __restrict__ p, long n16) {
  long i = (long)blockIdx.x * blockDim.x + threadIdx.x;
  if (i < n16) { v4u z = {0u, 0u, 0u, 0u}; p[i] = z; }
}

// f32 [K][N] -> bf16 [N][K] (transposed, so GEMM B-fragments are contiguous)
__global__ void k_tr_f32_to_bf16(const float* __restrict__ s, bf16* __restrict__ d, int K, int N) {
  int i = blockIdx.x * blockDim.x + threadIdx.x;
  if (i >= K * N) return;
  int k = i / N, n = i % N;
  d[n * K + k] = (bf16)s[i];
}

// enc3d conv1 weights (16,112,27) f32 -> (tap, co, ci=128) bf16, ci padded with 0
__global__ void k_pack_w3d0(const float* __restrict__ w, bf16* __restrict__ d) {
  int i = blockIdx.x * blockDim.x + threadIdx.x;
  if (i >= 27 * 16 * 128) return;
  int tap = i >> 11;              // /2048
  int q   = i & 2047;
  int co  = q >> 7, ci = q & 127;
  d[i] = (ci < 112) ? (bf16)w[(co * 112 + ci) * 27 + tap] : (bf16)0.0f;
}

// codebook -> bf16 copy + per-row squared norm
__global__ void k_cb_prep(const float* __restrict__ cb, bf16* __restrict__ cbb, float* __restrict__ cbn) {
  __shared__ float red[256];
  int k = blockIdx.x, j = threadIdx.x;
  float v = cb[k * 256 + j];
  cbb[k * 256 + j] = (bf16)v;
  red[j] = v * v; __syncthreads();
  for (int s = 128; s > 0; s >>= 1) { if (j < s) red[j] += red[j + s]; __syncthreads(); }
  if (j == 0) cbn[k] = red[0];
}

// ---------------------------------------------------------------- enc2d fused (both layers in LDS)
// Output layout: haloed channels-last (n, dd=9, hh=7, ww=15, ci=128); halo pre-zeroed by k_memset16.
__global__ void __launch_bounds__(256) k_enc2d(
    const float* __restrict__ x,
    const float* __restrict__ w1, const float* __restrict__ b1,
    const float* __restrict__ g1, const float* __restrict__ bb1,
    const float* __restrict__ m1, const float* __restrict__ v1,
    const float* __restrict__ w2, const float* __restrict__ b2,
    const float* __restrict__ g2, const float* __restrict__ bb2,
    const float* __restrict__ m2, const float* __restrict__ v2,
    bf16* __restrict__ A2) {
  __shared__ float xin[25 * 50];      // 1250
  __shared__ float h1[8 * 13 * 25];   // 2600
  int s   = blockIdx.x;               // ((b*5+nc)*7+r)*5+c
  int b   = s / 175;
  int nc  = (s / 35) % 5;
  int r   = (s / 5) % 7;
  int c   = s % 5;
  int n3  = s / 35;                   // b*5+nc
  int tid = threadIdx.x;
  long xbase = (long)b * 218750 + (long)r * 1250 + (long)c * 250 + nc * 50;
  for (int i = tid; i < 1250; i += 256) {
    int f = i / 50, t = i % 50;
    xin[i] = x[xbase + (long)f * 8750 + t];
  }
  __syncthreads();
  // layer 1: (1,25,50) -> (8,13,25), BN+SiLU
  for (int o = tid; o < 2600; o += 256) {
    int co = o / 325, p = o % 325, i = p / 25, j = p % 25;
    float sum = 0.0f;
    #pragma unroll
    for (int ki = 0; ki < 3; ++ki) {
      int ii = 2 * i - 1 + ki; if (ii < 0 || ii >= 25) continue;
      #pragma unroll
      for (int kj = 0; kj < 3; ++kj) {
        int jj = 2 * j - 1 + kj; if (jj < 0 || jj >= 50) continue;
        sum += xin[ii * 50 + jj] * w1[co * 9 + ki * 3 + kj];
      }
    }
    float sc = g1[co] * rsqrtf(v1[co] + EPS_BN);
    float sh = (b1[co] - m1[co]) * sc + bb1[co];
    h1[o] = silu_f(sum * sc + sh);
  }
  __syncthreads();
  // layer 2: (8,13,25) -> (16,7,13), BN+SiLU
  // interior store: (d,h,w) = (r,c,j) -> indices (r+1, c+1, j+1) in haloed volume
  for (int o = tid; o < 1456; o += 256) {
    int co = o / 91, p = o % 91, i = p / 13, j = p % 13;
    float sum = 0.0f;
    for (int ci = 0; ci < 8; ++ci) {
      const float* wp = &w2[(co * 8 + ci) * 9];
      const float* hp = &h1[ci * 325];
      #pragma unroll
      for (int ki = 0; ki < 3; ++ki) {
        int ii = 2 * i - 1 + ki; if (ii < 0 || ii >= 13) continue;
        #pragma unroll
        for (int kj = 0; kj < 3; ++kj) {
          int jj = 2 * j - 1 + kj; if (jj < 0 || jj >= 25) continue;
          sum += hp[ii * 25 + jj] * wp[ki * 3 + kj];
        }
      }
    }
    float sc = g2[co] * rsqrtf(v2[co] + EPS_BN);
    float sh = (b2[co] - m2[co]) * sc + bb2[co];
    float y  = silu_f(sum * sc + sh);
    A2[((((long)n3 * 9 + (r + 1)) * 7 + (c + 1)) * 15 + (j + 1)) * 128 + co * 7 + i] = (bf16)y;
  }
}

// ---------------------------------------------------------------- enc3d conv1 via WMMA implicit GEMM
// M = 640*84 positions, N = 16 channels, K = 128ci * 27 taps (bf16, f32 acc)
// A2 haloed channels-last (n,9,7,15,128): every (tap,k-step) is a constant offset from one
// per-lane base pointer -> pure global_load_b128 + WMMA inner loop, software pipelined.
__global__ void __launch_bounds__(128) k_e3d1_wmma(
    const bf16* __restrict__ A2, const bf16* __restrict__ Wb,
    const float* __restrict__ bconv, const float* __restrict__ g,
    const float* __restrict__ bb, const float* __restrict__ bm,
    const float* __restrict__ bv, float* __restrict__ A3) {
  int wave = threadIdx.x >> 5, lane = threadIdx.x & 31;
  int tile = blockIdx.x * 4 + wave;       // 3360 tiles total
  int half = lane >> 4;
  int res  = lane & 15;
  int mr   = tile * 16 + res;             // gather row for this lane
  int n    = mr / 84, pos = mr % 84;
  int od = pos / 21, oh = (pos % 21) / 7, ow = pos % 7;
  // base at (dd,hh,ww) = (2*od, 2*oh, 2*ow); tap (kd,kh,kw) adds constant (kd*105+kh*15+kw)*128
  const bf16* ap  = A2 + ((((long)n * 9 + 2 * od) * 7 + 2 * oh) * 15 + 2 * ow) * 128;
  const bf16* wb0 = Wb + res * 128;

  auto load_step = [&](int t, v16bf& a, v16bf& bfr) {
    int tap = t >> 2, kb = (t & 3) * 32;
    int kd = tap / 9, kh = (tap / 3) % 3, kw = tap % 3;
    int toff = (kd * 105 + kh * 15 + kw) * 128;
    a = frag16(ap + toff + kb + 8 * half, ap + toff + kb + 16 + 8 * half);
    const bf16* wp = wb0 + tap * 2048;
    bfr = frag16(wp + kb + 8 * half, wp + kb + 16 + 8 * half);
  };

  v8f acc = {0.f, 0.f, 0.f, 0.f, 0.f, 0.f, 0.f, 0.f};
  v16bf a_c, b_c;
  load_step(0, a_c, b_c);
  #pragma unroll 4
  for (int t = 0; t < 107; ++t) {           // 27 taps * 4 k-steps = 108 total
    v16bf a_n, b_n;
    load_step(t + 1, a_n, b_n);             // prefetch next step's fragments
    acc = __builtin_amdgcn_wmma_f32_16x16x32_bf16(false, a_c, false, b_c, (short)0, acc, false, false);
    a_c = a_n; b_c = b_n;
  }
  acc = __builtin_amdgcn_wmma_f32_16x16x32_bf16(false, a_c, false, b_c, (short)0, acc, false, false);

  int co = res;
  float sc = g[co] * rsqrtf(bv[co] + EPS_BN);
  float sh = (bconv[co] - bm[co]) * sc + bb[co];
  #pragma unroll
  for (int i = 0; i < 8; ++i) {
    int m  = tile * 16 + i + 8 * half;
    int nn = m / 84, pp = m % 84;
    A3[(long)(nn * 16 + co) * 84 + pp] = silu_f(acc[i] * sc + sh);
  }
}

// ---------------------------------------------------------------- enc3d conv2/conv3 (small, direct)
__global__ void k_e3d2(const float* __restrict__ A3, const float* __restrict__ w,
                       const float* __restrict__ bconv, const float* __restrict__ g,
                       const float* __restrict__ bb, const float* __restrict__ bm,
                       const float* __restrict__ bv, float* __restrict__ A4) {
  int o = blockIdx.x * blockDim.x + threadIdx.x;
  if (o >= 640 * 64 * 16) return;
  int n = o >> 10, q = o & 1023, co = q >> 4, p = q & 15;
  int od = p >> 3, oh = (p >> 2) & 1, ow = p & 3;
  float sum = 0.0f;
  for (int ci = 0; ci < 16; ++ci) {
    const float* ap = &A3[(long)(n * 16 + ci) * 84];
    const float* wp = &w[(co * 16 + ci) * 27];
    #pragma unroll
    for (int kd = 0; kd < 3; ++kd) {
      int d = 2 * od - 1 + kd; if (d < 0 || d >= 4) continue;
      #pragma unroll
      for (int kh = 0; kh < 3; ++kh) {
        int h = 2 * oh - 1 + kh; if (h < 0 || h >= 3) continue;
        #pragma unroll
        for (int kw = 0; kw < 3; ++kw) {
          int ww = 2 * ow - 1 + kw; if (ww < 0 || ww >= 7) continue;
          sum += ap[d * 21 + h * 7 + ww] * wp[(kd * 3 + kh) * 3 + kw];
        }
      }
    }
  }
  float sc = g[co] * rsqrtf(bv[co] + EPS_BN);
  float sh = (bconv[co] - bm[co]) * sc + bb[co];
  A4[o] = silu_f(sum * sc + sh);       // layout (n,64,2,2,4) == o
}

__global__ void k_e3d3(const float* __restrict__ A4, const float* __restrict__ w,
                       const float* __restrict__ bconv, const float* __restrict__ g,
                       const float* __restrict__ bb, const float* __restrict__ bm,
                       const float* __restrict__ bv, bf16* __restrict__ hflat) {
  int o = blockIdx.x * blockDim.x + threadIdx.x;
  if (o >= 640 * 128 * 2) return;
  int n = o >> 8, q = o & 255, co = q >> 1, ow = q & 1;
  float sum = 0.0f;
  for (int ci = 0; ci < 64; ++ci) {
    const float* ap = &A4[(long)(n * 64 + ci) * 16];
    const float* wp = &w[(co * 64 + ci) * 27];
    #pragma unroll
    for (int kd = 1; kd < 3; ++kd) {
      int d = kd - 1;
      #pragma unroll
      for (int kh = 1; kh < 3; ++kh) {
        int h = kh - 1;
        #pragma unroll
        for (int kw = 0; kw < 3; ++kw) {
          int ww = 2 * ow - 1 + kw; if (ww < 0 || ww >= 4) continue;
          sum += ap[d * 8 + h * 4 + ww] * wp[(kd * 3 + kh) * 3 + kw];
        }
      }
    }
  }
  float sc = g[co] * rsqrtf(bv[co] + EPS_BN);
  float sh = (bconv[co] - bm[co]) * sc + bb[co];
  hflat[n * 256 + co * 2 + ow] = (bf16)silu_f(sum * sc + sh);
}

// ---------------------------------------------------------------- bf16 WMMA GEMM, B stored [N][K], pipelined
__global__ void __launch_bounds__(128) k_gemm(const bf16* __restrict__ A, const bf16* __restrict__ Bt,
                                              const float* __restrict__ bias, float* __restrict__ C,
                                              int M, int N, int K, int act) {
  int wave = threadIdx.x >> 5, lane = threadIdx.x & 31;
  int mt = blockIdx.x;
  int nt = blockIdx.y * 4 + wave;
  int half = lane >> 4, res = lane & 15;
  const bf16* ra = A  + (long)(mt * 16 + res) * K;
  const bf16* rb = Bt + (long)(nt * 16 + res) * K;
  v8f acc = {0.f, 0.f, 0.f, 0.f, 0.f, 0.f, 0.f, 0.f};
  v16bf a_c = frag16(ra + 8 * half, ra + 16 + 8 * half);
  v16bf b_c = frag16(rb + 8 * half, rb + 16 + 8 * half);
  for (int kb = 32; kb < K; kb += 32) {
    v16bf a_n = frag16(ra + kb + 8 * half, ra + kb + 16 + 8 * half);
    v16bf b_n = frag16(rb + kb + 8 * half, rb + kb + 16 + 8 * half);
    acc = __builtin_amdgcn_wmma_f32_16x16x32_bf16(false, a_c, false, b_c, (short)0, acc, false, false);
    a_c = a_n; b_c = b_n;
  }
  acc = __builtin_amdgcn_wmma_f32_16x16x32_bf16(false, a_c, false, b_c, (short)0, acc, false, false);
  #pragma unroll
  for (int i = 0; i < 8; ++i) {
    int m = mt * 16 + i + 8 * half;
    int n = nt * 16 + res;
    float v = acc[i] + bias[n];
    if (act) v = silu_f(v);
    C[(long)m * N + n] = v;
  }
}

// ---------------------------------------------------------------- LayerNorm (row) + bf16 copy + row norms
__global__ void k_ln(const float* __restrict__ z, const float* __restrict__ g,
                     const float* __restrict__ b, float* __restrict__ zln,
                     bf16* __restrict__ zlnb, float* __restrict__ znorm) {
  __shared__ float red[256];
  int n = blockIdx.x, t = threadIdx.x;
  float v = z[n * 256 + t];
  red[t] = v; __syncthreads();
  for (int s = 128; s > 0; s >>= 1) { if (t < s) red[t] += red[t + s]; __syncthreads(); }
  float mu = red[0] * (1.0f / 256.0f); __syncthreads();
  float dv = v - mu;
  red[t] = dv * dv; __syncthreads();
  for (int s = 128; s > 0; s >>= 1) { if (t < s) red[t] += red[t + s]; __syncthreads(); }
  float var = red[0] * (1.0f / 256.0f); __syncthreads();
  float y = dv * rsqrtf(var + EPS_LN) * g[t] + b[t];
  zln[n * 256 + t] = y;
  zlnb[n * 256 + t] = (bf16)y;
  red[t] = y * y; __syncthreads();
  for (int s = 128; s > 0; s >>= 1) { if (t < s) red[t] += red[t + s]; __syncthreads(); }
  if (t == 0) znorm[n] = red[0];
}

// ---------------------------------------------------------------- VQ: fused score-WMMA + online argmin
__global__ void __launch_bounds__(32) k_vq(const bf16* __restrict__ Z, const bf16* __restrict__ CB,
                                           const float* __restrict__ zn, const float* __restrict__ cbn,
                                           int* __restrict__ idxi, float* __restrict__ idxf,
                                           float* __restrict__ counts) {
  __shared__ float sd[16][16];
  __shared__ int   sk[16][16];
  int lane = threadIdx.x & 31;
  int half = lane >> 4, res = lane & 15;
  int mt = blockIdx.x;
  const bf16* zr = Z + (long)(mt * 16 + res) * 256;
  v16bf afr[8];
  #pragma unroll
  for (int ks = 0; ks < 8; ++ks)
    afr[ks] = frag16(zr + ks * 32 + 8 * half, zr + ks * 32 + 16 + 8 * half);
  float bd[8]; int bk[8]; float zs[8];
  #pragma unroll
  for (int i = 0; i < 8; ++i) { bd[i] = 3.4e38f; bk[i] = 0; zs[i] = zn[mt * 16 + i + 8 * half]; }
  for (int ct = 0; ct < 32; ++ct) {
    const bf16* cr = CB + (long)(ct * 16 + res) * 256;
    v8f acc = {0.f, 0.f, 0.f, 0.f, 0.f, 0.f, 0.f, 0.f};
    #pragma unroll
    for (int ks = 0; ks < 8; ++ks) {
      v16bf bf = frag16(cr + ks * 32 + 8 * half, cr + ks * 32 + 16 + 8 * half);
      acc = __builtin_amdgcn_wmma_f32_16x16x32_bf16(false, afr[ks], false, bf, (short)0, acc, false, false);
    }
    float cn = cbn[ct * 16 + res];
    #pragma unroll
    for (int i = 0; i < 8; ++i) {
      float dist = zs[i] + cn - 2.0f * acc[i];
      if (dist < bd[i]) { bd[i] = dist; bk[i] = ct * 16 + res; }
    }
  }
  #pragma unroll
  for (int i = 0; i < 8; ++i) { sd[i + 8 * half][res] = bd[i]; sk[i + 8 * half][res] = bk[i]; }
  __syncthreads();
  if (lane < 16) {
    float best = sd[lane][0]; int kb = sk[lane][0];
    for (int c = 1; c < 16; ++c) {
      float dv = sd[lane][c]; int kv = sk[lane][c];
      if (dv < best || (dv == best && kv < kb)) { best = dv; kb = kv; }
    }
    int n = mt * 16 + lane;
    idxi[n] = kb;
    idxf[n] = (float)kb;
    atomicAdd(&counts[kb], 1.0f);
  }
}

// ---------------------------------------------------------------- losses / perplexity
__global__ void k_loss(const float* __restrict__ cb, const float* __restrict__ zln,
                       const int* __restrict__ idxi, float* __restrict__ sse) {
  __shared__ float red[256];
  int n = blockIdx.x, j = threadIdx.x;
  float e = cb[idxi[n] * 256 + j] - zln[n * 256 + j];
  red[j] = e * e; __syncthreads();
  for (int s = 128; s > 0; s >>= 1) { if (j < s) red[j] += red[j + s]; __syncthreads(); }
  if (j == 0) atomicAdd(sse, red[0]);
}

__global__ void k_final(const float* __restrict__ counts, const float* __restrict__ sse,
                        float* __restrict__ out) {
  __shared__ float red[512];
  int k = threadIdx.x;
  float p = counts[k] * (1.0f / 640.0f);
  red[k] = p * logf(p + 1e-10f);
  __syncthreads();
  for (int s = 256; s > 0; s >>= 1) { if (k < s) red[k] += red[k + s]; __syncthreads(); }
  if (k == 0) {
    float el = sse[0] * (1.0f / 163840.0f);   // e_loss == q_loss (forward value)
    out[1024640] = 1.25f * el;                // vq_loss = q + BETA*e
    out[1024641] = el;
    out[1024642] = el;
    out[1024643] = expf(-red[0]);             // perplexity
  }
}

// ---------------------------------------------------------------- decoder (tiny, direct)
__global__ void k_ct1(const float* __restrict__ CBact, const int* __restrict__ idxi,
                      const float* __restrict__ w, const float* __restrict__ bconv,
                      const float* __restrict__ g, const float* __restrict__ bb,
                      const float* __restrict__ bm, const float* __restrict__ bv,
                      float* __restrict__ D1) {
  int o = blockIdx.x * blockDim.x + threadIdx.x;
  if (o >= 640 * 32 * 11) return;
  int n = o / 352, q = o % 352, co = q / 11, t = q % 11;
  const float* xr = &CBact[(long)idxi[n] * 384];
  float sum = 0.0f;
  #pragma unroll
  for (int k = 0; k < 3; ++k) {
    int num = t + 1 - k;
    if (num < 0 || (num & 1)) continue;
    int l = num >> 1; if (l >= 6) continue;
    for (int ci = 0; ci < 64; ++ci) sum += xr[ci * 6 + l] * w[(ci * 32 + co) * 3 + k];
  }
  float sc = g[co] * rsqrtf(bv[co] + EPS_BN);
  float sh = (bconv[co] - bm[co]) * sc + bb[co];
  D1[o] = silu_f(sum * sc + sh);
}

__global__ void k_ct2(const float* __restrict__ D1, const float* __restrict__ w,
                      const float* __restrict__ bconv, const float* __restrict__ g,
                      const float* __restrict__ bb, const float* __restrict__ bm,
                      const float* __restrict__ bv, float* __restrict__ D2) {
  int o = blockIdx.x * blockDim.x + threadIdx.x;
  if (o >= 640 * 16 * 21) return;
  int n = o / 336, q = o % 336, co = q / 21, t = q % 21;
  const float* xr = &D1[(long)n * 352];
  float sum = 0.0f;
  #pragma unroll
  for (int k = 0; k < 3; ++k) {
    int num = t + 1 - k;
    if (num < 0 || (num & 1)) continue;
    int l = num >> 1; if (l >= 11) continue;
    for (int ci = 0; ci < 32; ++ci) sum += xr[ci * 11 + l] * w[(ci * 16 + co) * 3 + k];
  }
  float sc = g[co] * rsqrtf(bv[co] + EPS_BN);
  float sh = (bconv[co] - bm[co]) * sc + bb[co];
  D2[o] = silu_f(sum * sc + sh);
}

__device__ float ct3_eval(const float* __restrict__ x, const float* __restrict__ w,
                          float bsum, int co, int t) {
  float s = bsum;
  #pragma unroll
  for (int k = 0; k < 4; ++k) {
    int num = t + 1 - k;
    if (num < 0 || (num & 1)) continue;
    int l = num >> 1; if (l >= 21) continue;
    for (int ci = 0; ci < 16; ++ci) s += x[ci * 21 + l] * w[(ci * 32 + co) * 4 + k];
  }
  return s;
}

__global__ void k_ct3_resize(const float* __restrict__ D2, const float* __restrict__ w,
                             const float* __restrict__ b, float* __restrict__ out) {
  int o = blockIdx.x * blockDim.x + threadIdx.x;
  if (o >= 640 * 32 * 50) return;
  int n = o / 1600, q = o % 1600, co = q / 50, t = q % 50;
  const float* xr = &D2[(long)n * 336];
  float src = (t + 0.5f) * 0.84f - 0.5f;     // 42/50 scale, half-pixel
  float tf = floorf(src);
  float fr = src - tf;
  int t0 = (int)tf;
  int i0 = iclampi(t0, 0, 41), i1 = iclampi(t0 + 1, 0, 41);
  float v0 = ct3_eval(xr, w, b[co], co, i0);
  float v1 = (i1 == i0) ? v0 : ct3_eval(xr, w, b[co], co, i1);
  out[o] = v0 + fr * (v1 - v0);
}

// ---------------------------------------------------------------- launcher
extern "C" void kernel_launch(void* const* d_in, const int* in_sizes, int n_in,
                              void* d_out, int out_size, void* d_ws, size_t ws_size,
                              hipStream_t stream) {
  (void)in_sizes; (void)n_in; (void)out_size; (void)ws_size;
  const float* x       = (const float*)d_in[0];
  const float* e2d0_w  = (const float*)d_in[1];
  const float* e2d0_b  = (const float*)d_in[2];
  const float* e2d0_g  = (const float*)d_in[3];
  const float* e2d0_bb = (const float*)d_in[4];
  const float* e2d0_m  = (const float*)d_in[5];
  const float* e2d0_v  = (const float*)d_in[6];
  const float* e2d1_w  = (const float*)d_in[7];
  const float* e2d1_b  = (const float*)d_in[8];
  const float* e2d1_g  = (const float*)d_in[9];
  const float* e2d1_bb = (const float*)d_in[10];
  const float* e2d1_m  = (const float*)d_in[11];
  const float* e2d1_v  = (const float*)d_in[12];
  const float* e3d0_w  = (const float*)d_in[13];
  const float* e3d0_b  = (const float*)d_in[14];
  const float* e3d0_g  = (const float*)d_in[15];
  const float* e3d0_bb = (const float*)d_in[16];
  const float* e3d0_m  = (const float*)d_in[17];
  const float* e3d0_v  = (const float*)d_in[18];
  const float* e3d1_w  = (const float*)d_in[19];
  const float* e3d1_b  = (const float*)d_in[20];
  const float* e3d1_g  = (const float*)d_in[21];
  const float* e3d1_bb = (const float*)d_in[22];
  const float* e3d1_m  = (const float*)d_in[23];
  const float* e3d1_v  = (const float*)d_in[24];
  const float* e3d2_w  = (const float*)d_in[25];
  const float* e3d2_b  = (const float*)d_in[26];
  const float* e3d2_g  = (const float*)d_in[27];
  const float* e3d2_bb = (const float*)d_in[28];
  const float* e3d2_m  = (const float*)d_in[29];
  const float* e3d2_v  = (const float*)d_in[30];
  const float* proj_w  = (const float*)d_in[31];
  const float* proj_b  = (const float*)d_in[32];
  const float* ln_g    = (const float*)d_in[33];
  const float* ln_b    = (const float*)d_in[34];
  const float* cb      = (const float*)d_in[35];
  const float* dpw     = (const float*)d_in[36];
  const float* dpb     = (const float*)d_in[37];
  const float* dc0_w   = (const float*)d_in[38];
  const float* dc0_b   = (const float*)d_in[39];
  const float* dc0_g   = (const float*)d_in[40];
  const float* dc0_bb  = (const float*)d_in[41];
  const float* dc0_m   = (const float*)d_in[42];
  const float* dc0_v   = (const float*)d_in[43];
  const float* dc1_w   = (const float*)d_in[44];
  const float* dc1_b   = (const float*)d_in[45];
  const float* dc1_g   = (const float*)d_in[46];
  const float* dc1_bb  = (const float*)d_in[47];
  const float* dc1_m   = (const float*)d_in[48];
  const float* dc1_v   = (const float*)d_in[49];
  const float* lw      = (const float*)d_in[50];
  const float* lb      = (const float*)d_in[51];
  float* out = (float*)d_out;

  char* wsp = (char*)d_ws;
  size_t off = 0;
  auto take = [&](size_t bytes) -> void* {
    size_t a = (off + 255) & ~(size_t)255;
    off = a + bytes;
    return (void*)(wsp + a);
  };
  const long A2_ELEMS = (long)640 * 9 * 7 * 15 * 128;         // haloed channels-last
  bf16*  A2     = (bf16*) take((size_t)A2_ELEMS * 2);
  bf16*  Wc1    = (bf16*) take((size_t)27 * 16 * 128 * 2);    // (tap, co, ci)
  float* A3     = (float*)take((size_t)640 * 16 * 84 * 4);
  float* A4     = (float*)take((size_t)640 * 64 * 16 * 4);
  bf16*  hflat  = (bf16*) take((size_t)640 * 256 * 2);
  bf16*  projwT = (bf16*) take((size_t)256 * 256 * 2);        // [N][K]
  float* z      = (float*)take((size_t)640 * 256 * 4);
  float* zln    = (float*)take((size_t)640 * 256 * 4);
  bf16*  zlnb   = (bf16*) take((size_t)640 * 256 * 2);
  float* znorm  = (float*)take((size_t)640 * 4);
  bf16*  cbb    = (bf16*) take((size_t)512 * 256 * 2);
  float* cbn    = (float*)take((size_t)512 * 4);
  float* counts = (float*)take((size_t)513 * 4);              // [512] = sse accumulator
  float* sse    = counts + 512;
  int*   idxi   = (int*)  take((size_t)640 * 4);
  bf16*  decwT  = (bf16*) take((size_t)384 * 256 * 2);        // [N][K]
  float* CBact  = (float*)take((size_t)512 * 384 * 4);
  float* D1     = (float*)take((size_t)640 * 32 * 11 * 4);
  float* D2     = (float*)take((size_t)640 * 16 * 21 * 4);

  // prep / conversions
  long n16 = A2_ELEMS * 2 / 16;                               // 16B chunks
  k_memset16<<<(unsigned)((n16 + 255) / 256), 256, 0, stream>>>((v4u*)A2, n16);
  k_tr_f32_to_bf16<<<256, 256, 0, stream>>>(proj_w, projwT, 256, 256);
  k_tr_f32_to_bf16<<<384, 256, 0, stream>>>(dpw, decwT, 256, 384);
  k_pack_w3d0<<<216, 256, 0, stream>>>(e3d0_w, Wc1);
  k_cb_prep<<<512, 256, 0, stream>>>(cb, cbb, cbn);
  k_zero<<<3, 256, 0, stream>>>(counts, 513);

  // encoder
  k_enc2d<<<22400, 256, 0, stream>>>(x, e2d0_w, e2d0_b, e2d0_g, e2d0_bb, e2d0_m, e2d0_v,
                                     e2d1_w, e2d1_b, e2d1_g, e2d1_bb, e2d1_m, e2d1_v, A2);
  k_e3d1_wmma<<<840, 128, 0, stream>>>(A2, Wc1, e3d0_b, e3d0_g, e3d0_bb, e3d0_m, e3d0_v, A3);
  k_e3d2<<<2560, 256, 0, stream>>>(A3, e3d1_w, e3d1_b, e3d1_g, e3d1_bb, e3d1_m, e3d1_v, A4);
  k_e3d3<<<640, 256, 0, stream>>>(A4, e3d2_w, e3d2_b, e3d2_g, e3d2_bb, e3d2_m, e3d2_v, hflat);

  // projection + LN + VQ
  k_gemm<<<dim3(40, 4), 128, 0, stream>>>(hflat, projwT, proj_b, z, 640, 256, 256, 0);
  k_ln<<<640, 256, 0, stream>>>(z, ln_g, ln_b, zln, zlnb, znorm);
  k_vq<<<40, 32, 0, stream>>>(zlnb, cbb, znorm, cbn, idxi, out + 1024000, counts);
  k_loss<<<640, 256, 0, stream>>>(cb, zln, idxi, sse);
  k_final<<<1, 512, 0, stream>>>(counts, sse, out);

  // decoder: zq == codebook[idx]  =>  precompute silu(codebook @ dec_proj_w + b) once, gather per row
  k_gemm<<<dim3(32, 6), 128, 0, stream>>>(cbb, decwT, dpb, CBact, 512, 384, 256, 1);
  k_ct1<<<880, 256, 0, stream>>>(CBact, idxi, dc0_w, dc0_b, dc0_g, dc0_bb, dc0_m, dc0_v, D1);
  k_ct2<<<840, 256, 0, stream>>>(D1, dc1_w, dc1_b, dc1_g, dc1_bb, dc1_m, dc1_v, D2);
  k_ct3_resize<<<4000, 256, 0, stream>>>(D2, lw, lb, out);
}
